// HDG_66623532695755
// MI455X (gfx1250) — matrix operand order repaired and checked by reading, not compiled, not run on hardware
//
#include <hip/hip_runtime.h>
#include <hip/hip_bf16.h>
#include <math.h>

// ---------------------------------------------------------------------------
// HDG forward for MI455X (gfx1250, wave32).
//  - Embedding table (51.2MB) is L2-resident (192MB L2): workload is
//    L2-gather-bound, so normalized embeddings are stored as f16 to halve
//    gather bytes, and the per-edge cosine dots run through
//    v_wmma_f32_16x16x32_f16 (16 edges/tile, diag of 16x16 accumulator).
//  - User->item direction is perfect CSR (DEG=16 contiguous edges/user):
//    atomic-free wave-per-user SpMM. Item direction uses f32 atomics.
//  - Reciprocals via v_rcp_f32 (__builtin_amdgcn_rcpf) instead of the IEEE
//    division chain: one TRANS op per sigmoid / degree-normalize.
// ---------------------------------------------------------------------------

typedef __attribute__((ext_vector_type(16))) _Float16 v16h;
typedef __attribute__((ext_vector_type(8)))  _Float16 v8h;
typedef __attribute__((ext_vector_type(2)))  _Float16 v2h;
typedef __attribute__((ext_vector_type(8)))  float    v8f;

#define NU     100000
#define NI     100000
#define NN     (NU + NI)          // 200000 nodes
#define DIM    64
#define DEG    16
#define EDGES  (NU * DEG)         // 1,600,000
#define USZ    (NU + 1)           // user_size = n_users + 1 (faithful)
#define EPS_N  1e-8f
#define EPS_D  1e-7f
#define PRUNE_T 0.05f

__device__ __forceinline__ float fast_rcp(float x) {
  return __builtin_amdgcn_rcpf(x);            // v_rcp_f32
}

// ---------------- normalize rows, write f16 ----------------
// one wave per row; lane handles 2 dims (float2)
__global__ void hdg_normalize_f16(const float* __restrict__ src,
                                  _Float16* __restrict__ normh) {
  int row  = blockIdx.x * 8 + (threadIdx.x >> 5);
  int lane = threadIdx.x & 31;
  if (row >= NN) return;
  float2 v = reinterpret_cast<const float2*>(src)[(size_t)row * 32 + lane];
  float ss = v.x * v.x + v.y * v.y;
  #pragma unroll
  for (int off = 16; off > 0; off >>= 1) ss += __shfl_xor(ss, off, 32);
  float inv = fast_rcp(fmaxf(sqrtf(ss), EPS_N));
  v2h h; h.x = (_Float16)(v.x * inv); h.y = (_Float16)(v.y * inv);
  reinterpret_cast<v2h*>(normh)[(size_t)row * 32 + lane] = h;
}

// ---------------- per-edge cosine sims via WMMA ----------------
// One wave per 16-edge tile. A = a_norm rows of the 16 edges (16x64 f16),
// B = b_norm rows as columns (64x16 f16). Two chained 16x16x32 WMMAs (K=64);
// diagonal of the f32 accumulator is the per-edge dot. Then gate + prune.
//
// A fragment layout (16-bit A 16x32): lane L holds row M=L&15;
//   halves[0..7]  = K = 32c + (L>=16 ? 16 : 0) + 0..7
//   halves[8..15] = K = 32c + (L>=16 ? 16 : 0) + 8..15
// B fragment layout (16-bit B 32x16): lane L holds col N=L&15;
//   halves[0..15] = K = 32c + (L>=16 ? 16 : 0) + 0..15
__global__ void hdg_edge_sims_wmma(const _Float16* __restrict__ normh,
                                   const int* __restrict__ rowi,
                                   const int* __restrict__ coli,
                                   const float* __restrict__ gate_w,
                                   const float* __restrict__ gate_b,
                                   float* __restrict__ pruned) {
  const int lane = threadIdx.x & 31;
  const int tile = blockIdx.x * 8 + (threadIdx.x >> 5);
  if (tile >= EDGES / 16) return;               // grid sized exactly: full waves
  const int e0 = tile * 16;
  const int m  = lane & 15;
  const int hi = lane >> 4;

  const int eM = e0 + m;
  const int u  = rowi[eM];                      // user node for A-row m
  const int it = USZ + coli[eM];                // b_norm index (user_size offset)
  const _Float16* aRow = normh + (size_t)u  * DIM;
  const _Float16* bRow = normh + (size_t)it * DIM;

  v8f acc = {};
  #pragma unroll
  for (int c = 0; c < 2; ++c) {
    const int kA = c * 32 + hi * 16;
    v8h alo = *reinterpret_cast<const v8h*>(aRow + kA);
    v8h ahi = *reinterpret_cast<const v8h*>(aRow + kA + 8);
    v16h a;
    #pragma unroll
    for (int i = 0; i < 8; ++i) { a[i] = alo[i]; a[i + 8] = ahi[i]; }
    const int kB = c * 32 + hi * 16;
    v16h b = *reinterpret_cast<const v16h*>(bRow + kB);
    acc = __builtin_amdgcn_wmma_f32_16x16x32_f16(
        /*neg_a=*/false, a, /*neg_b=*/false, b,
        /*c_mod=*/(short)0, acc, /*reuse_a=*/false, /*reuse_b=*/false);
  }

  // diagonal (j,j): lanes 0..7 hold M=r (r=VGPR idx), lanes 16..31 hold M=8+r
  float d = 0.0f; int j = -1;
  if (lane < 8)        { j = lane;      d = acc[lane];      }
  else if (lane >= 24) { j = lane - 16; d = acc[lane - 24]; }
  if (j >= 0) {
    const float gw = gate_w[0];
    const float gb = gate_b[0];
    float s = (d + 1.0f) * 0.5f;
    float g = fast_rcp(1.0f + __expf(-(s * gw + gb)));   // sigmoid via v_rcp
    float p = s * g;
    p = (p < PRUNE_T) ? 0.0f : p;
    pruned[e0 + j] = p;
  }
}

// ---------------- degree: user rows (contiguous CSR), item rows init ----------
__global__ void hdg_degree_base(const float* __restrict__ pruned,
                                float* __restrict__ deg) {
  int i = blockIdx.x * blockDim.x + threadIdx.x;
  if (i >= NN) return;
  if (i < NU) {
    const float4* p4 = reinterpret_cast<const float4*>(pruned + (size_t)i * DEG);
    float s = 0.0f;
    #pragma unroll
    for (int k = 0; k < DEG / 4; ++k) {
      float4 v = p4[k];
      s += v.x + v.y + v.z + v.w;
    }
    deg[i] = s + EPS_D;
  } else {
    deg[i] = EPS_D;                               // item side filled by atomics
  }
}

__global__ void hdg_degree_scatter(const float* __restrict__ pruned,
                                   const int* __restrict__ coli,
                                   float* __restrict__ deg) {
  int e = blockIdx.x * blockDim.x + threadIdx.x;
  if (e >= EDGES) return;
  float p = pruned[e];
  if (p != 0.0f) atomicAdd(&deg[NU + coli[e]], p);   // adjacency offset = n_users
}

// ---------------- zero item rows of dst (atomics accumulate into them) -------
__global__ void hdg_zero_items(float* __restrict__ dst) {
  long i = (long)blockIdx.x * blockDim.x + threadIdx.x;     // float4 index
  if (i >= (long)NI * (DIM / 4)) return;
  reinterpret_cast<float4*>(dst + (size_t)NU * DIM)[i] = make_float4(0, 0, 0, 0);
}

// ---------------- SpMM direction 1: users gather items (no atomics) ----------
__global__ void hdg_spmm_user(const float* __restrict__ src,
                              const float* __restrict__ pruned,
                              const int* __restrict__ coli,
                              const float* __restrict__ deg,
                              float* __restrict__ dst) {
  int u    = blockIdx.x * 8 + (threadIdx.x >> 5);
  int lane = threadIdx.x & 31;
  if (u >= NU) return;
  const int*   cp = coli   + (size_t)u * DEG;
  const float* wp = pruned + (size_t)u * DEG;
  float2 acc = make_float2(0.0f, 0.0f);
  #pragma unroll 4
  for (int k = 0; k < DEG; ++k) {
    float w = wp[k];
    if (k + 1 < DEG)                               // L2 warm-up for next row
      __builtin_prefetch(src + ((size_t)NU + cp[k + 1]) * DIM, 0, 1);
    if (w != 0.0f) {
      size_t node = (size_t)NU + cp[k];
      float2 v = reinterpret_cast<const float2*>(src)[node * 32 + lane];
      acc.x += w * v.x;
      acc.y += w * v.y;
    }
  }
  float inv = fast_rcp(deg[u]);                    // deg >= 1e-7, safe for rcp
  reinterpret_cast<float2*>(dst)[(size_t)u * 32 + lane] =
      make_float2(acc.x * inv, acc.y * inv);
}

// ---------------- SpMM direction 2: items scatter from users (atomics) -------
__global__ void hdg_spmm_item(const float* __restrict__ src,
                              const float* __restrict__ pruned,
                              const int* __restrict__ rowi,
                              const int* __restrict__ coli,
                              const float* __restrict__ deg,
                              float* __restrict__ dst) {
  int e    = blockIdx.x * 8 + (threadIdx.x >> 5);   // one wave per edge
  int lane = threadIdx.x & 31;
  if (e >= EDGES) return;
  float p = pruned[e];
  if (p == 0.0f) return;                            // uniform across the wave
  size_t node = (size_t)NU + coli[e];
  float  w    = p * fast_rcp(deg[node]);
  float2 v = reinterpret_cast<const float2*>(src)[(size_t)rowi[e] * 32 + lane];
  float* dp = dst + node * DIM + lane * 2;
  atomicAdd(dp,     w * v.x);
  atomicAdd(dp + 1, w * v.y);
}

// ---------------- elementwise copy / accumulate / finalize ----------------
__global__ void hdg_copy(const float* __restrict__ a, float* __restrict__ o) {
  long i = (long)blockIdx.x * blockDim.x + threadIdx.x;
  if (i >= (long)NN * (DIM / 4)) return;
  reinterpret_cast<float4*>(o)[i] = reinterpret_cast<const float4*>(a)[i];
}
__global__ void hdg_accum(const float* __restrict__ a, float* __restrict__ o) {
  long i = (long)blockIdx.x * blockDim.x + threadIdx.x;
  if (i >= (long)NN * (DIM / 4)) return;
  float4 x = reinterpret_cast<float4*>(o)[i];
  float4 y = reinterpret_cast<const float4*>(a)[i];
  reinterpret_cast<float4*>(o)[i] = make_float4(x.x + y.x, x.y + y.y, x.z + y.z, x.w + y.w);
}
__global__ void hdg_finalize(const float* __restrict__ a, float* __restrict__ o) {
  long i = (long)blockIdx.x * blockDim.x + threadIdx.x;
  if (i >= (long)NN * (DIM / 4)) return;
  const float s = 1.0f / 3.0f;
  float4 x = reinterpret_cast<float4*>(o)[i];
  float4 y = reinterpret_cast<const float4*>(a)[i];
  reinterpret_cast<float4*>(o)[i] =
      make_float4((x.x + y.x) * s, (x.y + y.y) * s, (x.z + y.z) * s, (x.w + y.w) * s);
}

// ---------------------------------------------------------------------------
extern "C" void kernel_launch(void* const* d_in, const int* in_sizes, int n_in,
                              void* d_out, int out_size, void* d_ws, size_t ws_size,
                              hipStream_t stream) {
  const float* in_embs = (const float*)d_in[0];
  const float* gate_w  = (const float*)d_in[1];
  const float* gate_b  = (const float*)d_in[2];
  const int*   rowi    = (const int*)d_in[3];
  const int*   coli    = (const int*)d_in[4];
  float* out = (float*)d_out;

  // workspace carve-out (256B aligned slabs)
  char* ws = (char*)d_ws;
  size_t off = 0;
  auto carve = [&](size_t bytes) -> void* {
    void* p = ws + off;
    off = (off + bytes + 255) & ~(size_t)255;
    return p;
  };
  _Float16* normh = (_Float16*)carve((size_t)NN * DIM * sizeof(_Float16));
  float*    pruned = (float*)carve((size_t)EDGES * sizeof(float));
  float*    deg    = (float*)carve((size_t)NN * sizeof(float));
  float*    emb1   = (float*)carve((size_t)NN * DIM * sizeof(float));
  float*    emb2   = (float*)carve((size_t)NN * DIM * sizeof(float));

  const dim3 B256(256);
  const int rowsBlocks  = (NN + 7) / 8;                 // normalize: 8 rows/block
  const int tileBlocks  = (EDGES / 16) / 8;             // edge sims: 8 waves/block
  const int nodeBlocks  = (NN + 255) / 256;
  const int edgeBlocks  = (EDGES + 255) / 256;
  const int itemZBlocks = (int)(((long)NI * (DIM / 4) + 255) / 256);
  const int userBlocks  = (NU + 7) / 8;                 // spmm_user: 8 waves/block
  const int edgeWBlocks = (EDGES + 7) / 8;              // spmm_item: 8 waves/block
  const int elemBlocks  = (int)(((long)NN * (DIM / 4) + 255) / 256);

  // out = e0
  hdg_copy<<<elemBlocks, B256, 0, stream>>>(in_embs, out);

  const float* src = in_embs;
  float* dsts[2] = { emb1, emb2 };
  for (int layer = 0; layer < 2; ++layer) {
    float* dst = dsts[layer];
    hdg_normalize_f16 <<<rowsBlocks,  B256, 0, stream>>>(src, normh);
    hdg_edge_sims_wmma<<<tileBlocks,  B256, 0, stream>>>(normh, rowi, coli,
                                                         gate_w, gate_b, pruned);
    hdg_degree_base   <<<nodeBlocks,  B256, 0, stream>>>(pruned, deg);
    hdg_degree_scatter<<<edgeBlocks,  B256, 0, stream>>>(pruned, coli, deg);
    hdg_zero_items    <<<itemZBlocks, B256, 0, stream>>>(dst);
    hdg_spmm_user     <<<userBlocks,  B256, 0, stream>>>(src, pruned, coli, deg, dst);
    hdg_spmm_item     <<<edgeWBlocks, B256, 0, stream>>>(src, pruned, rowi, coli, deg, dst);
    if (layer == 0) hdg_accum   <<<elemBlocks, B256, 0, stream>>>(dst, out);
    else            hdg_finalize<<<elemBlocks, B256, 0, stream>>>(dst, out);
    src = dst;
  }
}
